// CrossViewAttention_90546500534784
// MI455X (gfx1250) — matrix operand order repaired and checked by reading, not compiled, not run on hardware
//
#include <hip/hip_runtime.h>
#include <hip/hip_bf16.h>
#include <math.h>
#include <stdint.h>

// ---------------------------------------------------------------------------
// Problem constants (from the reference)
// ---------------------------------------------------------------------------
#define BB     2
#define DIMC   256
#define HEADS  4
#define DHC    64
#define MIDC   64
#define MTOK   4096   // S*S  (S=64)
#define NTOK   4096   // H*W  (32*128)
#define MPPF   1.6f   // 0.2*512/64

typedef __bf16 bf16_t;
typedef __attribute__((ext_vector_type(16))) __bf16 v16bf;
typedef __attribute__((ext_vector_type(8)))  float  v8f;

// ---------------------------------------------------------------------------
// CDNA5 async global->LDS copy (ASYNCcnt path), per CDNA5 ISA.
// lds_off is the low-32-bit LDS byte offset of the destination (flat LDS
// addresses truncate to the LDS offset per the aperture rules).
// ---------------------------------------------------------------------------
__device__ __forceinline__ void async_copy_b128(void* lds_dst, const void* gsrc) {
  unsigned off = (unsigned)(uintptr_t)lds_dst;
  asm volatile("global_load_async_to_lds_b128 %0, %1, off"
               :: "v"(off), "v"(gsrc)
               : "memory");
}
__device__ __forceinline__ void wait_async0() {
  asm volatile("s_wait_asynccnt 0x0" ::: "memory");
}

// ---------------------------------------------------------------------------
// WMMA helpers (CDNA5: V_WMMA_F32_16X16X32_BF16)
// ---------------------------------------------------------------------------
__device__ __forceinline__ v8f wmma_bf16(v16bf a, v16bf b, v8f c) {
  return __builtin_amdgcn_wmma_f32_16x16x32_bf16(false, a, false, b,
                                                 (short)0, c, false, false);
}

// A-fragment: 16x32 (MxK), row-major source with leading dim ldm.
__device__ __forceinline__ void load_a_frag(v16bf& A, const bf16_t* base,
                                            int row0, int ldm, int k0, int lane) {
  int m  = lane & 15;
  int kb = (lane < 16) ? 0 : 8;
  const bf16_t* p = base + (size_t)(row0 + m) * (size_t)ldm;
#pragma unroll
  for (int j = 0; j < 8; ++j) {
    int k = k0 + ((j < 4) ? (kb + 2 * j) : (16 + kb + 2 * (j - 4)));
    A[2 * j]     = p[k];
    A[2 * j + 1] = p[k + 1];
  }
}

// B-fragment: 32x16 (KxN), row-major bf16 src: elem (k,n) at src[(k0+k)*ld+n0+n]
__device__ __forceinline__ void load_b_rm(v16bf& B, const bf16_t* src,
                                          int k0, int ld, int n0, int lane) {
  int kb = (lane < 16) ? 0 : 8;
  const bf16_t* p = src + (size_t)k0 * (size_t)ld + n0 + (lane & 15);
#pragma unroll
  for (int j = 0; j < 8; ++j) {
    int k = (j < 4) ? (kb + 2 * j) : (16 + kb + 2 * (j - 4));
    B[2 * j]     = p[(size_t)k * (size_t)ld];
    B[2 * j + 1] = p[(size_t)(k + 1) * (size_t)ld];
  }
}

// B-fragment of X^T, X row-major [n, d]: B[k=d_local, n] = X[(n0+n)*ld + d0+k]
__device__ __forceinline__ void load_bT(v16bf& B, const bf16_t* src,
                                        int d0, int ld, int n0, int lane) {
  int kb = (lane < 16) ? 0 : 8;
  const bf16_t* p = src + (size_t)(n0 + (lane & 15)) * (size_t)ld + d0;
#pragma unroll
  for (int j = 0; j < 8; ++j) {
    int k = (j < 4) ? (kb + 2 * j) : (16 + kb + 2 * (j - 4));
    B[2 * j]     = p[k];
    B[2 * j + 1] = p[k + 1];
  }
}

// ---------------------------------------------------------------------------
// 0) f32 -> bf16 weight conversion (run once per weight tensor)
// ---------------------------------------------------------------------------
__global__ __launch_bounds__(256) void cvt_bf16_kernel(
    const float* __restrict__ src, bf16_t* __restrict__ dst, int n) {
  int i = blockIdx.x * 256 + threadIdx.x;
  if (i < n) dst[i] = (bf16_t)src[i];
}

// ---------------------------------------------------------------------------
// 1) query prep: pos-embed + channel L2 norm + sat 1x1 conv + per-head LN
// ---------------------------------------------------------------------------
__global__ __launch_bounds__(256) void prep_query_kernel(
    const float* __restrict__ sat_emb, const float* __restrict__ bev_w,
    const float* __restrict__ bev_b, const float* __restrict__ sat_w,
    const float* __restrict__ sat_b, const float* __restrict__ q_g,
    const float* __restrict__ q_b, bf16_t* __restrict__ qn) {
  __shared__ float sA[256];
  __shared__ float sB[256];
  int c  = threadIdx.x;
  int bm = blockIdx.x;             // b*4096 + m
  int m  = bm & (MTOK - 1);
  int i  = m >> 6, j = m & 63;
  float gx = MPPF * (float)(i - 32);
  float gy = MPPF * (float)(j - 32);
  float pos = bev_w[2 * c] * gx + bev_w[2 * c + 1] * gy + bev_b[c];
  sA[c] = pos * pos;
  __syncthreads();
  for (int off = 128; off >= 1; off >>= 1) {
    if (c < off) sA[c] += sA[c + off];
    __syncthreads();
  }
  float nrm = fmaxf(sqrtf(sA[0]), 1e-12f);
  __syncthreads();
  float q = pos / nrm + sat_w[c] * sat_emb[bm] + sat_b[c];
  sA[c] = q;
  sB[c] = q * q;
  __syncthreads();
  for (int off = 32; off >= 1; off >>= 1) {
    if ((c & 63) < off) { sA[c] += sA[c + off]; sB[c] += sB[c + off]; }
    __syncthreads();
  }
  int   hb   = c & ~63;
  float mean = sA[hb] * (1.0f / 64.0f);
  float var  = sB[hb] * (1.0f / 64.0f) - mean * mean;
  float rstd = rsqrtf(var + 1e-5f);
  int   d    = c & 63;
  qn[(size_t)bm * DIMC + c] = (bf16_t)((q - mean) * rstd * q_g[d] + q_b[d]);
}

// ---------------------------------------------------------------------------
// 2) key prep: transpose grd_feat [B,C,N] -> keyf f32 [B,N,C]; per-head LN
// ---------------------------------------------------------------------------
__global__ __launch_bounds__(256) void prep_key_kernel(
    const float* __restrict__ grd_feat, const float* __restrict__ k_g,
    const float* __restrict__ k_b, const float* __restrict__ v_g,
    const float* __restrict__ v_b, float* __restrict__ keyf,
    bf16_t* __restrict__ kn, bf16_t* __restrict__ vn) {
  __shared__ float sA[256];
  __shared__ float sB[256];
  int c  = threadIdx.x;
  int bn = blockIdx.x;
  int b  = bn >> 12;
  int n  = bn & (NTOK - 1);
  float x = grd_feat[((size_t)b * DIMC + c) * NTOK + n];
  keyf[(size_t)bn * DIMC + c] = x;
  sA[c] = x;
  sB[c] = x * x;
  __syncthreads();
  for (int off = 32; off >= 1; off >>= 1) {
    if ((c & 63) < off) { sA[c] += sA[c + off]; sB[c] += sB[c + off]; }
    __syncthreads();
  }
  int   hb   = c & ~63;
  float mean = sA[hb] * (1.0f / 64.0f);
  float var  = sB[hb] * (1.0f / 64.0f) - mean * mean;
  float rstd = rsqrtf(var + 1e-5f);
  float xn   = (x - mean) * rstd;
  int   d    = c & 63;
  kn[(size_t)bn * DIMC + c] = (bf16_t)(xn * k_g[d] + k_b[d]);
  vn[(size_t)bn * DIMC + c] = (bf16_t)(xn * v_g[d] + v_b[d]);
}

// ---------------------------------------------------------------------------
// 3) Generic WMMA GEMM: C[M,N] = act(A[M,K](bf16) @ W[K,N](bf16) + bias)
//    128 threads / 4 waves, 64x64 tile; gridDim.z batches (z = b*heads + h)
// ---------------------------------------------------------------------------
__global__ __launch_bounds__(128) void gemm_wmma_kernel(
    const bf16_t* __restrict__ A, long aBase, long aStrB, long aStrH, int lda,
    const bf16_t* __restrict__ W, int ldw, const float* __restrict__ bias,
    float* __restrict__ Cf, bf16_t* __restrict__ Cb, long cBase, long cStrB,
    long cStrH, int ldc, const float* __restrict__ resid, int K, int heads,
    int doGelu) {
  int lane = threadIdx.x & 31;
  int w    = threadIdx.x >> 5;
  int z    = blockIdx.z;
  int b    = z / heads;
  int h    = z - b * heads;
  const bf16_t* Ap = A + aBase + (long)b * aStrB + (long)h * aStrH;
  long cOff = cBase + (long)b * cStrB + (long)h * cStrH;
  int  m0   = blockIdx.x * 64 + w * 16;
  int  n0   = blockIdx.y * 64;

  v8f Cacc[4];
#pragma unroll
  for (int t = 0; t < 4; ++t)
#pragma unroll
    for (int r = 0; r < 8; ++r) Cacc[t][r] = 0.0f;

  for (int k0 = 0; k0 < K; k0 += 32) {
    v16bf af;
    load_a_frag(af, Ap, m0, lda, k0, lane);
    v16bf bf;
#pragma unroll
    for (int t = 0; t < 4; ++t) {
      load_b_rm(bf, W, k0, ldw, n0 + 16 * t, lane);
      Cacc[t] = wmma_bf16(af, bf, Cacc[t]);
    }
  }

  int nl = lane & 15;
  int mh = (lane < 16) ? 0 : 8;
#pragma unroll
  for (int t = 0; t < 4; ++t) {
    int   nn = n0 + 16 * t + nl;
    float bi = bias ? bias[nn] : 0.0f;
#pragma unroll
    for (int r = 0; r < 8; ++r) {
      float v = Cacc[t][r] + bi;
      if (doGelu) v = 0.5f * v * (1.0f + erff(v * 0.70710678118f));
      long idx = cOff + (long)(m0 + mh + r) * ldc + nn;
      if (resid) v += resid[idx];
      if (Cf) Cf[idx] = v;
      if (Cb) Cb[idx] = (bf16_t)v;
    }
  }
}

// ---------------------------------------------------------------------------
// 4) Row LayerNorm over 256 channels (one wave per row)
// ---------------------------------------------------------------------------
__global__ __launch_bounds__(256) void ln_row_kernel(
    const float* __restrict__ X, const float* __restrict__ g,
    const float* __restrict__ bt, float* __restrict__ outF,
    bf16_t* __restrict__ outB, int rows) {
  int lane = threadIdx.x & 31;
  int w    = threadIdx.x >> 5;
  int row  = blockIdx.x * 8 + w;
  if (row >= rows) return;
  const float* x = X + (size_t)row * DIMC;
  float v[8], s = 0.0f, s2 = 0.0f;
#pragma unroll
  for (int i = 0; i < 8; ++i) {
    v[i] = x[lane + 32 * i];
    s += v[i];
    s2 += v[i] * v[i];
  }
#pragma unroll
  for (int msk = 1; msk < 32; msk <<= 1) {
    s  += __shfl_xor(s, msk, 32);
    s2 += __shfl_xor(s2, msk, 32);
  }
  float mean = s * (1.0f / DIMC);
  float var  = s2 * (1.0f / DIMC) - mean * mean;
  float rstd = rsqrtf(var + 1e-5f);
#pragma unroll
  for (int i = 0; i < 8; ++i) {
    int   c = lane + 32 * i;
    float y = (v[i] - mean) * rstd * g[c] + bt[c];
    if (outF) outF[(size_t)row * DIMC + c] = y;
    if (outB) outB[(size_t)row * DIMC + c] = (bf16_t)y;
  }
}

// ---------------------------------------------------------------------------
// 5) Flash cross-attention, per (b, h). Block = 256 threads (8 waves),
//    128 query rows per block. K/V staged 64 keys at a time into LDS via
//    CDNA5 async global->LDS copies; softmax is online in registers.
// ---------------------------------------------------------------------------
#define PLD 40   // padded LDS leading dim for P tiles
#define KCH 64   // keys per staged chunk
__global__ __launch_bounds__(256) void flash_attn_kernel(
    const bf16_t* __restrict__ qp, const bf16_t* __restrict__ kp,
    const bf16_t* __restrict__ vp, bf16_t* __restrict__ aout) {
  __shared__ bf16_t kt[KCH * 64];           // [n_local][d]  (8 KB)
  __shared__ bf16_t vt[KCH * 64];           // [n_local][d]  (8 KB)
  __shared__ bf16_t pbuf[8][16 * PLD];      // per-wave P    (10 KB)
  int tid  = threadIdx.x;
  int lane = tid & 31;
  int w    = tid >> 5;
  int mt   = blockIdx.x & 31;
  int bh   = blockIdx.x >> 5;
  int h    = bh & (HEADS - 1);
  int b    = bh >> 2;
  int h64  = h * 64;
  long qbase = (long)b * MTOK * DIMC + h64;
  long kbase = (long)b * NTOK * DIMC + h64;
  int  row0  = mt * 128 + w * 16;

  const bf16_t* ksrc = kp + kbase;
  const bf16_t* vsrc = vp + kbase;

  v16bf qf0, qf1;
  load_a_frag(qf0, qp + qbase, row0, DIMC, 0, lane);
  load_a_frag(qf1, qp + qbase, row0, DIMC, 32, lane);

  v8f Oacc[4];
  float mrow[8], lrow[8];
#pragma unroll
  for (int r = 0; r < 8; ++r) { mrow[r] = -1e30f; lrow[r] = 0.0f; }
#pragma unroll
  for (int t = 0; t < 4; ++t)
#pragma unroll
    for (int r = 0; r < 8; ++r) Oacc[t][r] = 0.0f;

  const float scale = 0.125f;  // DH^-0.5
  bf16_t* P  = &pbuf[w][0];
  int     nl = lane & 15;
  int     mh = (lane < 16) ? 0 : 8;
  int     srow = tid >> 3;        // 0..31  (staging row)
  int     sseg = (tid & 7) * 8;   // staging column (8 bf16 = 16 B)

  for (int n0 = 0; n0 < NTOK; n0 += KCH) {
    // ---- cooperative async stage of K/V chunk into LDS ----
    const bf16_t* kg = ksrc + (size_t)(n0 + srow) * DIMC + sseg;
    const bf16_t* vg = vsrc + (size_t)(n0 + srow) * DIMC + sseg;
    async_copy_b128(&kt[srow * 64 + sseg], kg);
    async_copy_b128(&kt[(srow + 32) * 64 + sseg], kg + 32 * DIMC);
    async_copy_b128(&vt[srow * 64 + sseg], vg);
    async_copy_b128(&vt[(srow + 32) * 64 + sseg], vg + 32 * DIMC);
    wait_async0();
    __syncthreads();

#pragma unroll
    for (int sc = 0; sc < 2; ++sc) {
      int nn = sc * 32;  // sub-chunk base within LDS tile
      v8f s0, s1;
#pragma unroll
      for (int r = 0; r < 8; ++r) { s0[r] = 0.0f; s1[r] = 0.0f; }
      v16bf bf;
      load_bT(bf, kt, 0, 64, nn, lane);       s0 = wmma_bf16(qf0, bf, s0);
      load_bT(bf, kt, 32, 64, nn, lane);      s0 = wmma_bf16(qf1, bf, s0);
      load_bT(bf, kt, 0, 64, nn + 16, lane);  s1 = wmma_bf16(qf0, bf, s1);
      load_bT(bf, kt, 32, 64, nn + 16, lane); s1 = wmma_bf16(qf1, bf, s1);

#pragma unroll
      for (int r = 0; r < 8; ++r) {
        float a0 = s0[r] * scale, a1 = s1[r] * scale;
        float cm = fmaxf(a0, a1);
#pragma unroll
        for (int msk = 1; msk < 16; msk <<= 1)
          cm = fmaxf(cm, __shfl_xor(cm, msk, 32));
        float mnew  = fmaxf(mrow[r], cm);
        float alpha = __expf(mrow[r] - mnew);
        float p0 = __expf(a0 - mnew), p1 = __expf(a1 - mnew);
        float cs = p0 + p1;
#pragma unroll
        for (int msk = 1; msk < 16; msk <<= 1) cs += __shfl_xor(cs, msk, 32);
        mrow[r] = mnew;
        lrow[r] = lrow[r] * alpha + cs;
#pragma unroll
        for (int t = 0; t < 4; ++t) Oacc[t][r] *= alpha;
        P[(mh + r) * PLD + nl]      = (bf16_t)p0;
        P[(mh + r) * PLD + 16 + nl] = (bf16_t)p1;
      }
      __builtin_amdgcn_wave_barrier();
      asm volatile("" ::: "memory");

      v16bf pf;
      load_a_frag(pf, P, 0, PLD, 0, lane);
      v16bf vf;
#pragma unroll
      for (int t = 0; t < 4; ++t) {
        load_b_rm(vf, vt, nn, 64, 16 * t, lane);
        Oacc[t] = wmma_bf16(pf, vf, Oacc[t]);
      }
      __builtin_amdgcn_wave_barrier();
      asm volatile("" ::: "memory");
    }
    __syncthreads();
  }

#pragma unroll
  for (int r = 0; r < 8; ++r) {
    float inv  = 1.0f / lrow[r];
    long  base = (long)(b * MTOK + row0 + mh + r) * DIMC + h64;
#pragma unroll
    for (int t = 0; t < 4; ++t)
      aout[base + 16 * t + nl] = (bf16_t)(Oacc[t][r] * inv);
  }
}

// ---------------------------------------------------------------------------
// 6) Height attention: out[b,m] = softmax(hq@hk^T / 8) @ val (scalar value);
//    hk staged through LDS with async copies like the flash kernel.
// ---------------------------------------------------------------------------
__global__ __launch_bounds__(256) void height_attn_kernel(
    const bf16_t* __restrict__ hq, const bf16_t* __restrict__ hk,
    const float* __restrict__ val, float* __restrict__ out) {
  __shared__ bf16_t kt[KCH * 64];   // hk chunk  (8 KB)
  int tid  = threadIdx.x;
  int lane = tid & 31;
  int w    = tid >> 5;
  int mt   = blockIdx.x & 31;
  int b    = blockIdx.x >> 5;
  int row0 = mt * 128 + w * 16;
  long qbase = (long)b * MTOK * MIDC;
  long kbase = (long)b * NTOK * MIDC;

  v16bf qf0, qf1;
  load_a_frag(qf0, hq + qbase, row0, MIDC, 0, lane);
  load_a_frag(qf1, hq + qbase, row0, MIDC, 32, lane);

  float mrow[8], lrow[8], acc[8];
#pragma unroll
  for (int r = 0; r < 8; ++r) { mrow[r] = -1e30f; lrow[r] = 0.0f; acc[r] = 0.0f; }

  const float scale = 0.125f;  // MID^-0.5
  int nl = lane & 15;
  int mh = (lane < 16) ? 0 : 8;
  int srow = tid >> 3;
  int sseg = (tid & 7) * 8;
  const bf16_t* ksrc = hk + kbase;

  for (int n0 = 0; n0 < NTOK; n0 += KCH) {
    const bf16_t* kg = ksrc + (size_t)(n0 + srow) * MIDC + sseg;
    async_copy_b128(&kt[srow * 64 + sseg], kg);
    async_copy_b128(&kt[(srow + 32) * 64 + sseg], kg + 32 * MIDC);
    wait_async0();
    __syncthreads();
    if (n0 + KCH < NTOK)  // prefetch next chunk of values (global_prefetch_b8)
      __builtin_prefetch(val + (size_t)b * NTOK + n0 + KCH + nl, 0, 0);

#pragma unroll
    for (int sc = 0; sc < 2; ++sc) {
      int nn = sc * 32;
      v8f s0, s1;
#pragma unroll
      for (int r = 0; r < 8; ++r) { s0[r] = 0.0f; s1[r] = 0.0f; }
      v16bf bf;
      load_bT(bf, kt, 0, 64, nn, lane);       s0 = wmma_bf16(qf0, bf, s0);
      load_bT(bf, kt, 32, 64, nn, lane);      s0 = wmma_bf16(qf1, bf, s0);
      load_bT(bf, kt, 0, 64, nn + 16, lane);  s1 = wmma_bf16(qf0, bf, s1);
      load_bT(bf, kt, 32, 64, nn + 16, lane); s1 = wmma_bf16(qf1, bf, s1);

      float v0 = val[(size_t)b * NTOK + n0 + nn + nl];
      float v1 = val[(size_t)b * NTOK + n0 + nn + 16 + nl];
#pragma unroll
      for (int r = 0; r < 8; ++r) {
        float a0 = s0[r] * scale, a1 = s1[r] * scale;
        float cm = fmaxf(a0, a1);
#pragma unroll
        for (int msk = 1; msk < 16; msk <<= 1)
          cm = fmaxf(cm, __shfl_xor(cm, msk, 32));
        float mnew  = fmaxf(mrow[r], cm);
        float alpha = __expf(mrow[r] - mnew);
        float p0 = __expf(a0 - mnew), p1 = __expf(a1 - mnew);
        float cs = p0 + p1;
        float ca = p0 * v0 + p1 * v1;
#pragma unroll
        for (int msk = 1; msk < 16; msk <<= 1) {
          cs += __shfl_xor(cs, msk, 32);
          ca += __shfl_xor(ca, msk, 32);
        }
        mrow[r] = mnew;
        lrow[r] = lrow[r] * alpha + cs;
        acc[r]  = acc[r] * alpha + ca;
      }
    }
    __syncthreads();
  }

  if (nl == 0) {
#pragma unroll
    for (int r = 0; r < 8; ++r)
      out[(size_t)b * MTOK + row0 + mh + r] = acc[r] / lrow[r];
  }
}

// ---------------------------------------------------------------------------
// Host orchestration
// ---------------------------------------------------------------------------
extern "C" void kernel_launch(void* const* d_in, const int* in_sizes, int n_in,
                              void* d_out, int out_size, void* d_ws,
                              size_t ws_size, hipStream_t stream) {
  const float* sat_emb    = (const float*)d_in[0];
  const float* grd_feat   = (const float*)d_in[1];
  const float* grd_height = (const float*)d_in[2];
  const float* bev_w = (const float*)d_in[4];
  const float* bev_b = (const float*)d_in[5];
  const float* sat_w = (const float*)d_in[6];
  const float* sat_b = (const float*)d_in[7];
  const float* q_g = (const float*)d_in[8],  *q_b = (const float*)d_in[9];
  const float* q_w = (const float*)d_in[10], *q_wb = (const float*)d_in[11];
  const float* k_g = (const float*)d_in[12], *k_b = (const float*)d_in[13];
  const float* k_w = (const float*)d_in[14], *k_wb = (const float*)d_in[15];
  const float* v_g = (const float*)d_in[16], *v_b = (const float*)d_in[17];
  const float* v_w = (const float*)d_in[18], *v_wb = (const float*)d_in[19];
  const float* proj_w = (const float*)d_in[20], *proj_b = (const float*)d_in[21];
  const float* pre_g = (const float*)d_in[22], *pre_b = (const float*)d_in[23];
  const float* mlp_w1 = (const float*)d_in[24], *mlp_b1 = (const float*)d_in[25];
  const float* mlp_w2 = (const float*)d_in[26], *mlp_b2 = (const float*)d_in[27];
  const float* post_g = (const float*)d_in[28], *post_b = (const float*)d_in[29];
  const float* hq_g = (const float*)d_in[30], *hq_b = (const float*)d_in[31];
  const float* hq_w = (const float*)d_in[32], *hq_wb = (const float*)d_in[33];
  const float* hk_g = (const float*)d_in[34], *hk_b = (const float*)d_in[35];
  const float* hk_w = (const float*)d_in[36], *hk_wb = (const float*)d_in[37];

  // Workspace layout (aliased across pipeline phases; peak ~45 MiB)
  char* ws = (char*)d_ws;
  const size_t MB = 1024 * 1024;
  bf16_t* qn     = (bf16_t*)(ws + 0 * MB);   // 4 MiB  (later: zln)
  float*  keyf   = (float*)(ws + 4 * MB);    // 8 MiB
  bf16_t* kn     = (bf16_t*)(ws + 12 * MB);  // 4 MiB  (later: hql)
  bf16_t* vn     = (bf16_t*)(ws + 16 * MB);  // 4 MiB  (later: hkl)
  bf16_t* qp     = (bf16_t*)(ws + 20 * MB);  // 4 MiB \  (later: zproj/z2 8 MiB)
  bf16_t* kp     = (bf16_t*)(ws + 24 * MB);  // 4 MiB /
  bf16_t* vp     = (bf16_t*)(ws + 28 * MB);  // 4 MiB \  (later: zpre/z3 8 MiB)
  bf16_t* attn_a = (bf16_t*)(ws + 32 * MB);  // 4 MiB /
  float*  zproj  = (float*)(ws + 20 * MB);
  float*  zpre   = (float*)(ws + 28 * MB);
  bf16_t* zln    = (bf16_t*)(ws + 0 * MB);
  bf16_t* hid    = (bf16_t*)(ws + 36 * MB);  // 8 MiB  (later: hqo/hko)
  float*  z2     = (float*)(ws + 20 * MB);
  float*  z3     = (float*)(ws + 28 * MB);
  bf16_t* hql    = (bf16_t*)(ws + 12 * MB);
  bf16_t* hkl    = (bf16_t*)(ws + 16 * MB);
  bf16_t* hqo    = (bf16_t*)(ws + 36 * MB);  // 1 MiB
  bf16_t* hko    = (bf16_t*)(ws + 37 * MB);  // 1 MiB

  // bf16 weight copies (converted once) at 44 MiB
  bf16_t* wbase = (bf16_t*)(ws + 44 * MB);
  bf16_t* wq    = wbase;                //  64*64
  bf16_t* wk    = wq + 4096;
  bf16_t* wv    = wk + 4096;
  bf16_t* wproj = wv + 4096;            // 256*256
  bf16_t* wm1   = wproj + 65536;        // 256*512
  bf16_t* wm2   = wm1 + 131072;         // 512*256
  bf16_t* whq   = wm2 + 131072;         // 256*64
  bf16_t* whk   = whq + 16384;          // 256*64

  const long MD   = (long)MTOK * DIMC;  // per-batch token*dim stride
  const int  ROWS = BB * MTOK;          // 8192

  // 0) one-time weight conversion to bf16
  cvt_bf16_kernel<<<16, 256, 0, stream>>>(q_w, wq, 4096);
  cvt_bf16_kernel<<<16, 256, 0, stream>>>(k_w, wk, 4096);
  cvt_bf16_kernel<<<16, 256, 0, stream>>>(v_w, wv, 4096);
  cvt_bf16_kernel<<<256, 256, 0, stream>>>(proj_w, wproj, 65536);
  cvt_bf16_kernel<<<512, 256, 0, stream>>>(mlp_w1, wm1, 131072);
  cvt_bf16_kernel<<<512, 256, 0, stream>>>(mlp_w2, wm2, 131072);
  cvt_bf16_kernel<<<64, 256, 0, stream>>>(hq_w, whq, 16384);
  cvt_bf16_kernel<<<64, 256, 0, stream>>>(hk_w, whk, 16384);

  // 1) prep
  prep_query_kernel<<<BB * MTOK, 256, 0, stream>>>(sat_emb, bev_w, bev_b,
                                                   sat_w, sat_b, q_g, q_b, qn);
  prep_key_kernel<<<BB * NTOK, 256, 0, stream>>>(grd_feat, k_g, k_b, v_g, v_b,
                                                 keyf, kn, vn);

  // 2) per-head q/k/v projections (batched over B*H): [4096,64] @ [64,64]
  dim3 gQKV(MTOK / 64, 1, BB * HEADS);
  gemm_wmma_kernel<<<gQKV, 128, 0, stream>>>(qn, 0, MD, 64, DIMC, wq, 64,
                                             q_wb, nullptr, qp, 0, MD, 64,
                                             DIMC, nullptr, 64, HEADS, 0);
  gemm_wmma_kernel<<<gQKV, 128, 0, stream>>>(kn, 0, MD, 64, DIMC, wk, 64,
                                             k_wb, nullptr, kp, 0, MD, 64,
                                             DIMC, nullptr, 64, HEADS, 0);
  gemm_wmma_kernel<<<gQKV, 128, 0, stream>>>(vn, 0, MD, 64, DIMC, wv, 64,
                                             v_wb, nullptr, vp, 0, MD, 64,
                                             DIMC, nullptr, 64, HEADS, 0);

  // 3) flash attention -> attn_a (bf16, heads concatenated)
  flash_attn_kernel<<<BB * HEADS * 32, 256, 0, stream>>>(qp, kp, vp, attn_a);

  // 4) output projection -> zproj (f32)
  dim3 gProj(ROWS / 64, DIMC / 64, 1);
  gemm_wmma_kernel<<<gProj, 128, 0, stream>>>(attn_a, 0, 0, 0, DIMC, wproj,
                                              DIMC, proj_b, zproj, nullptr, 0,
                                              0, 0, DIMC, nullptr, DIMC, 1, 0);

  // 5) pre-LN -> zpre (f32) + zln (bf16)
  ln_row_kernel<<<ROWS / 8, 256, 0, stream>>>(zproj, pre_g, pre_b, zpre, zln,
                                              ROWS);

  // 6) MLP: 256 -> 512 (GELU) -> 256 (+ residual zpre) -> z2
  dim3 gM1(ROWS / 64, 512 / 64, 1);
  gemm_wmma_kernel<<<gM1, 128, 0, stream>>>(zln, 0, 0, 0, DIMC, wm1, 512,
                                            mlp_b1, nullptr, hid, 0, 0, 0, 512,
                                            nullptr, DIMC, 1, 1);
  dim3 gM2(ROWS / 64, DIMC / 64, 1);
  gemm_wmma_kernel<<<gM2, 128, 0, stream>>>(hid, 0, 0, 0, 512, wm2, DIMC,
                                            mlp_b2, z2, nullptr, 0, 0, 0, DIMC,
                                            zpre, 512, 1, 0);

  // 7) post-LN -> z3 (f32)
  ln_row_kernel<<<ROWS / 8, 256, 0, stream>>>(z2, post_g, post_b, z3, nullptr,
                                              ROWS);

  // 8) height-attention LNs + projections
  ln_row_kernel<<<ROWS / 8, 256, 0, stream>>>(z3, hq_g, hq_b, nullptr, hql,
                                              ROWS);
  ln_row_kernel<<<ROWS / 8, 256, 0, stream>>>(keyf, hk_g, hk_b, nullptr, hkl,
                                              ROWS);
  dim3 gH(ROWS / 64, 1, 1);
  gemm_wmma_kernel<<<gH, 128, 0, stream>>>(hql, 0, 0, 0, DIMC, whq, MIDC,
                                           hq_wb, nullptr, hqo, 0, 0, 0, MIDC,
                                           nullptr, DIMC, 1, 0);
  gemm_wmma_kernel<<<gH, 128, 0, stream>>>(hkl, 0, 0, 0, DIMC, whk, MIDC,
                                           hk_wb, nullptr, hko, 0, 0, 0, MIDC,
                                           nullptr, DIMC, 1, 0);

  // 9) height attention -> d_out [B,1,64,64]
  height_attn_kernel<<<BB * 32, 256, 0, stream>>>(hqo, hko, grd_height,
                                                  (float*)d_out);
}
